// SparseMLP_29686813950055
// MI455X (gfx1250) — matrix operand order repaired and checked by reading, not compiled, not run on hardware
//
#include <hip/hip_runtime.h>
#include <hip/hip_bf16.h>
#include <math.h>

// ---------------------------------------------------------------------------
// MoE top-1 sparse MLP for MI455X (gfx1250, wave32, WMMA bf16 path).
// T=8192 tokens, H=I=2048, E=8 experts, capacity C=1280.
// 172 GFLOP in the expert GEMMs vs ~500MB of traffic: bf16 WMMA (16x16x32,
// fp32 accum) keeps compute at parity with the 23.3 TB/s HBM roofline;
// fp32 WMMA (16x16x4) would be 8x off the matrix ceiling.
// GEMM: 128x128 tile, BK=32, double-buffered LDS, async-to-LDS DMA for the
// A tile, XOR-swizzled LDS to kill ds_load_b128 bank conflicts.
// ---------------------------------------------------------------------------

#define TOK 8192
#define HID 2048
#define FFN 2048
#define NE  8
#define CAP 1280   // floor(1.25*8192/8)=1280, even, >=4

typedef __attribute__((ext_vector_type(16))) __bf16 v16bf;
typedef __attribute__((ext_vector_type(8)))  float  v8f;

union Frag { v16bf v; uint4 q[2]; };
union U128 { uint4 q; unsigned short s[8]; };

#if defined(__gfx1250__) && __has_builtin(__builtin_amdgcn_global_load_async_to_lds_b128)
#define USE_ASYNC_LDS 1
// builtin takes pointers to 16-byte int vectors: global src, LDS dst
typedef int v4i_vs __attribute__((__vector_size__(4 * sizeof(int))));
typedef __attribute__((address_space(1))) v4i_vs AsyncGT;
typedef __attribute__((address_space(3))) v4i_vs AsyncLT;
#else
#define USE_ASYNC_LDS 0
#endif

static __device__ __forceinline__ void wait_async() {
#if USE_ASYNC_LDS
#if __has_builtin(__builtin_amdgcn_s_wait_asynccnt)
  __builtin_amdgcn_s_wait_asynccnt(0);
#else
  asm volatile("s_wait_asynccnt 0x0" ::: "memory");
#endif
#endif
}

static __device__ __forceinline__ unsigned short f2bf(float x) {
  return __builtin_bit_cast(unsigned short, (__bf16)x);
}

// ---------------------------------------------------------------------------
// fp32 -> bf16 bulk conversion (8 elements / thread, b128 in / b128 out)
// ---------------------------------------------------------------------------
__global__ void cvt_bf16_kernel(const float* __restrict__ src,
                                unsigned short* __restrict__ dst,
                                long long n) {
  long long i = ((long long)blockIdx.x * blockDim.x + threadIdx.x) * 8;
  if (i + 8 > n) return;
  const float4* s = (const float4*)(src + i);
  float4 a = s[0], b = s[1];
  U128 o;
  o.s[0] = f2bf(a.x); o.s[1] = f2bf(a.y); o.s[2] = f2bf(a.z); o.s[3] = f2bf(a.w);
  o.s[4] = f2bf(b.x); o.s[5] = f2bf(b.y); o.s[6] = f2bf(b.z); o.s[7] = f2bf(b.w);
  *(uint4*)(dst + i) = o.q;
}

// ---------------------------------------------------------------------------
// Gate: logits = tokens @ gate_w^T (fp32), softmax, top-1 id + top prob.
// ---------------------------------------------------------------------------
__global__ void gate_kernel(const float* __restrict__ tokens,
                            const float* __restrict__ gw,
                            int* __restrict__ top1,
                            float* __restrict__ ptop) {
  __shared__ float lgw[NE * 512];
  int t = blockIdx.x * 256 + threadIdx.x;
  float acc[NE];
#pragma unroll
  for (int e = 0; e < NE; ++e) acc[e] = 0.f;

  for (int h0 = 0; h0 < HID; h0 += 512) {
    for (int i = threadIdx.x; i < NE * 512 / 4; i += 256) {
      int e = i >> 7, j = i & 127;
      ((float4*)lgw)[i] = *(const float4*)(gw + (size_t)e * HID + h0 + j * 4);
    }
    __syncthreads();
    const float4* tk = (const float4*)(tokens + (size_t)t * HID + h0);
    for (int j = 0; j < 128; ++j) {
      float4 x = tk[j];
#pragma unroll
      for (int e = 0; e < NE; ++e) {
        float4 g = ((const float4*)lgw)[e * 128 + j];
        acc[e] += x.x * g.x + x.y * g.y + x.z * g.z + x.w * g.w;
      }
    }
    __syncthreads();
  }

  int am = 0; float mx = acc[0];
#pragma unroll
  for (int e = 1; e < NE; ++e) { if (acc[e] > mx) { mx = acc[e]; am = e; } }
  float s = 0.f;
#pragma unroll
  for (int e = 0; e < NE; ++e) s += expf(acc[e] - mx);
  top1[t] = am;
  ptop[t] = 1.0f / s;
}

__global__ void init_slots_kernel(int* __restrict__ token_of,
                                  float* __restrict__ weight_of) {
  int i = blockIdx.x * blockDim.x + threadIdx.x;
  if (i < NE * CAP) { token_of[i] = -1; weight_of[i] = 0.f; }
}

// ---------------------------------------------------------------------------
// Ordered per-expert rank (token-order cumsum), capacity drop, inverse maps.
// ---------------------------------------------------------------------------
__global__ void rank_kernel(const int* __restrict__ top1,
                            const float* __restrict__ ptop,
                            int* __restrict__ token_of,
                            float* __restrict__ weight_of) {
  __shared__ int scan[1024];
  int tid = threadIdx.x;
  int base[NE];
#pragma unroll
  for (int e = 0; e < NE; ++e) base[e] = 0;

  for (int c0 = 0; c0 < TOK; c0 += 1024) {
    int t = c0 + tid;
    int e = top1[t];
    float p = ptop[t];
    int rank = -1;
#pragma unroll 1
    for (int ex = 0; ex < NE; ++ex) {
      int flag = (e == ex) ? 1 : 0;
      scan[tid] = flag;
      __syncthreads();
      for (int off = 1; off < 1024; off <<= 1) {
        int v = (tid >= off) ? scan[tid - off] : 0;
        __syncthreads();
        scan[tid] += v;
        __syncthreads();
      }
      if (flag) rank = base[ex] + scan[tid] - 1;
      int tot = scan[1023];
      __syncthreads();
      base[ex] += tot;
    }
    if (rank >= 0 && rank < CAP) {
      token_of[e * CAP + rank]  = t;
      weight_of[e * CAP + rank] = p;
    }
  }
}

// ---------------------------------------------------------------------------
// Dispatch gather: xd[e][c][:] = bf16(tokens[token_of[e][c]][:]) or zeros.
// ---------------------------------------------------------------------------
__global__ void dispatch_kernel(const float* __restrict__ tokens,
                                const int* __restrict__ token_of,
                                unsigned short* __restrict__ xd) {
  int slot = blockIdx.x;              // e*CAP + c
  int t = token_of[slot];
  int base = threadIdx.x * 8;
  U128 o;
  if (t >= 0) {
    const float4* s = (const float4*)(tokens + (size_t)t * HID + base);
    float4 a = s[0], b = s[1];
    o.s[0] = f2bf(a.x); o.s[1] = f2bf(a.y); o.s[2] = f2bf(a.z); o.s[3] = f2bf(a.w);
    o.s[4] = f2bf(b.x); o.s[5] = f2bf(b.y); o.s[6] = f2bf(b.z); o.s[7] = f2bf(b.w);
  } else {
    o.q = make_uint4(0u, 0u, 0u, 0u);
  }
  *(uint4*)(xd + (size_t)slot * HID + base) = o.q;
}

// ---------------------------------------------------------------------------
// bf16 WMMA GEMM, one expert per blockIdx.z.
//   MODE 0: Hmid = bf16(gelu(A @ B))
//   MODE 1: out[token_of[R]] = weight_of[R] * (A @ B)[R]
// 128x128 tile, BK=32, 8 waves (2x4), 4x2 accumulators of 16x16 per wave.
// Double-buffered LDS; A tile via GLOBAL_LOAD_ASYNC_TO_LDS_B128, B tile
// transposed into [n][k] with packed b32 stores. XOR swizzle on 16B chunks
// spreads fragment ds_load_b128 across banks.
// ---------------------------------------------------------------------------
template <int MODE>
__global__ __launch_bounds__(256) void moe_gemm_kernel(
    const unsigned short* __restrict__ Aall,
    const unsigned short* __restrict__ Ball,
    unsigned short* __restrict__ HmidOut,
    float* __restrict__ out,
    const int* __restrict__ token_of,
    const float* __restrict__ weight_of,
    int Mrows, int Kdim, int Ncols) {
  __shared__ unsigned short lsA[2][128 * 32];   // [row][k], chunk-swizzled
  __shared__ unsigned short lsB[2][128 * 32];   // transposed [n][k], swizzled

  const int tid  = threadIdx.x;
  const int lane = tid & 31;
  const int wid  = tid >> 5;
  const int wrow = (wid >> 2) * 64;          // 0 / 64
  const int wcol = (wid & 3) * 32;           // 0 / 32 / 64 / 96
  const int half = lane >> 4;
  const int l15  = lane & 15;

  const int rowBase = blockIdx.y * 128;
  const int colBase = blockIdx.x * 128;
  const int e = blockIdx.z;

  const unsigned short* A  = Aall + (size_t)e * Mrows * Kdim;
  const unsigned short* Bm = Ball + (size_t)e * Kdim * Ncols;

  // loop-invariant tile-copy geometry -------------------------------------
  const int aRow0 = tid >> 2,           aGrp0 = tid & 3;
  const int aRow1 = aRow0 + 64,         aGrp1 = aGrp0;
  const size_t aG0 = (size_t)(rowBase + aRow0) * Kdim + aGrp0 * 8;
  const size_t aG1 = (size_t)(rowBase + aRow1) * Kdim + aGrp1 * 8;
  const int aL0 = aRow0 * 32 + ((aGrp0 ^ ((aRow0 >> 2) & 3)) * 8);
  const int aL1 = aRow1 * 32 + ((aGrp1 ^ ((aRow1 >> 2) & 3)) * 8);

  const int bK = (tid >> 4) * 2;             // even k-row pair within tile
  const int bN = (tid & 15) * 8;             // n-group start
  const size_t bG = (size_t)bK * Ncols + colBase + bN;

  uint4 aReg[2]; U128 bReg[2];

  auto issueA = [&](int k0, int buf) {
#if USE_ASYNC_LDS
    __builtin_amdgcn_global_load_async_to_lds_b128(
        (AsyncGT*)(A + aG0 + k0), (AsyncLT*)&lsA[buf][aL0], 0, 0);
    __builtin_amdgcn_global_load_async_to_lds_b128(
        (AsyncGT*)(A + aG1 + k0), (AsyncLT*)&lsA[buf][aL1], 0, 0);
#else
    (void)buf;
    aReg[0] = *(const uint4*)(A + aG0 + k0);
    aReg[1] = *(const uint4*)(A + aG1 + k0);
#endif
  };
  auto loadB = [&](int k0) {
    bReg[0].q = *(const uint4*)(Bm + bG + (size_t)k0 * Ncols);
    bReg[1].q = *(const uint4*)(Bm + bG + (size_t)(k0 + 1) * Ncols);
  };
  auto commit = [&](int buf) {
#if !USE_ASYNC_LDS
    *(uint4*)&lsA[buf][aL0] = aReg[0];
    *(uint4*)&lsA[buf][aL1] = aReg[1];
#endif
    const int chunk = bK >> 3;
#pragma unroll
    for (int j = 0; j < 8; ++j) {
      int n = bN + j;
      unsigned int v = (unsigned int)bReg[0].s[j] |
                       ((unsigned int)bReg[1].s[j] << 16);
      *(unsigned int*)&lsB[buf][n * 32 + ((chunk ^ ((n >> 2) & 3)) * 8) + (bK & 7)] = v;
    }
  };

  // ---- prologue: stage tile 0 -------------------------------------------
  issueA(0, 0);
  loadB(0);
  commit(0);
  wait_async();
  __syncthreads();

  v8f acc[4][2] = {};
  const int nIter = Kdim / 32;

  for (int it = 0; it < nIter; ++it) {
    const int cur = it & 1;
    const bool hasNext = (it + 1) < nIter;
    if (hasNext) {                 // start next tile's traffic early
      issueA((it + 1) * 32, cur ^ 1);
      loadB((it + 1) * 32);
    }

    // A fragment (ISA 7.12.2): lane<16 row M=l15 K={0..7,16..23}; lanes 16-31
    // K={8..15,24..31}. Two swizzled ds_load_b128 each.
    Frag a[4], b[2];
#pragma unroll
    for (int mt = 0; mt < 4; ++mt) {
      int row = wrow + mt * 16 + l15;
      int sw = (row >> 2) & 3;
      const unsigned short* base = &lsA[cur][row * 32];
      a[mt].q[0] = *(const uint4*)(base + ((half ^ sw) * 8));
      a[mt].q[1] = *(const uint4*)(base + (((2 + half) ^ sw) * 8));
    }
    // B fragment: lane holds column N=l15, K={0..7,16..23} (half 0) / {8..15,24..31} (half 1).
#pragma unroll
    for (int nt = 0; nt < 2; ++nt) {
      int n = wcol + nt * 16 + l15;
      int sw = (n >> 2) & 3;
      const unsigned short* base = &lsB[cur][n * 32];
      b[nt].q[0] = *(const uint4*)(base + ((half ^ sw) * 8));
      b[nt].q[1] = *(const uint4*)(base + (((2 + half) ^ sw) * 8));
    }
#pragma unroll
    for (int mt = 0; mt < 4; ++mt)
#pragma unroll
      for (int nt = 0; nt < 2; ++nt)
        acc[mt][nt] = __builtin_amdgcn_wmma_f32_16x16x32_bf16(
            false, a[mt].v, false, b[nt].v, (short)0, acc[mt][nt], false, false);

    if (hasNext) {
      commit(cur ^ 1);
      wait_async();
      __syncthreads();
    }
  }

  // C/D layout: VGPR r -> M = half*8 + r, N = l15
  if (MODE == 0) {
#pragma unroll
    for (int mt = 0; mt < 4; ++mt)
#pragma unroll
      for (int nt = 0; nt < 2; ++nt)
#pragma unroll
        for (int r = 0; r < 8; ++r) {
          int R  = rowBase + wrow + mt * 16 + half * 8 + r;
          int Cn = colBase + wcol + nt * 16 + l15;
          float x = acc[mt][nt][r];
          float g = 0.5f * x * (1.0f + erff(x * 0.70710678118654752f));
          HmidOut[(size_t)e * Mrows * Ncols + (size_t)R * Ncols + Cn] = f2bf(g);
        }
  } else {
#pragma unroll
    for (int mt = 0; mt < 4; ++mt) {
#pragma unroll
      for (int r = 0; r < 8; ++r) {
        int R = rowBase + wrow + mt * 16 + half * 8 + r;
        int t = token_of[e * CAP + R];
        if (t >= 0) {
          float w = weight_of[e * CAP + R];
#pragma unroll
          for (int nt = 0; nt < 2; ++nt) {
            int Cn = colBase + wcol + nt * 16 + l15;
            out[(size_t)t * Ncols + Cn] = w * acc[mt][nt][r];
          }
        }
      }
    }
  }
}

// ---------------------------------------------------------------------------
// host-side orchestration (graph-capture safe: only async ops on `stream`)
// ---------------------------------------------------------------------------
extern "C" void kernel_launch(void* const* d_in, const int* in_sizes, int n_in,
                              void* d_out, int out_size, void* d_ws, size_t ws_size,
                              hipStream_t stream) {
  const float* tokens = (const float*)d_in[0];  // [2,4096,2048]
  const float* gatew  = (const float*)d_in[1];  // [8,2048]
  const float* w1     = (const float*)d_in[2];  // [8,2048,2048]
  const float* w2     = (const float*)d_in[3];  // [8,2048,2048]
  float* out = (float*)d_out;                   // [2,4096,2048] fp32

  char* ws = (char*)d_ws;
  size_t off = 0;
  auto carve = [&](size_t bytes) -> void* {
    void* p = ws + off;
    off += (bytes + 255) & ~(size_t)255;
    return p;
  };
  unsigned short* w1b = (unsigned short*)carve((size_t)NE * HID * FFN * 2);
  unsigned short* w2b = (unsigned short*)carve((size_t)NE * FFN * HID * 2);
  unsigned short* xd  = (unsigned short*)carve((size_t)NE * CAP * HID * 2);
  unsigned short* hm  = (unsigned short*)carve((size_t)NE * CAP * FFN * 2);
  int*   top1      = (int*)carve((size_t)TOK * 4);
  float* ptop      = (float*)carve((size_t)TOK * 4);
  int*   token_of  = (int*)carve((size_t)NE * CAP * 4);
  float* weight_of = (float*)carve((size_t)NE * CAP * 4);
  (void)ws_size; (void)in_sizes; (void)n_in; (void)out_size;

  (void)hipMemsetAsync(d_out, 0, (size_t)TOK * HID * sizeof(float), stream);

  const long long nw = (long long)NE * HID * FFN;
  cvt_bf16_kernel<<<(int)(nw / (256 * 8)), 256, 0, stream>>>(w1, w1b, nw);
  cvt_bf16_kernel<<<(int)(nw / (256 * 8)), 256, 0, stream>>>(w2, w2b, nw);

  gate_kernel<<<TOK / 256, 256, 0, stream>>>(tokens, gatew, top1, ptop);
  init_slots_kernel<<<(NE * CAP + 255) / 256, 256, 0, stream>>>(token_of, weight_of);
  rank_kernel<<<1, 1024, 0, stream>>>(top1, ptop, token_of, weight_of);
  dispatch_kernel<<<NE * CAP, 256, 0, stream>>>(tokens, token_of, xd);

  dim3 grid1(FFN / 128, CAP / 128, NE);  // (16, 10, 8)
  moe_gemm_kernel<0><<<grid1, 256, 0, stream>>>(xd, w1b, hm, nullptr,
                                                token_of, weight_of, CAP, HID, FFN);
  dim3 grid2(HID / 128, CAP / 128, NE);
  moe_gemm_kernel<1><<<grid2, 256, 0, stream>>>(hm, w2b, nullptr, out,
                                                token_of, weight_of, CAP, FFN, HID);
}
